// MusicAttrSingleVAE_14963666059400
// MI455X (gfx1250) — matrix operand order repaired and checked by reading, not compiled, not run on hardware
//
#include <hip/hip_runtime.h>
#include <hip/hip_bf16.h>

// ---------------------------------------------------------------------------
// MusicAttrSingleVAE on gfx1250: bf16 WMMA GEMMs for every matmul, fp32 state.
// B=128, T=256, R=130, H=1024, Z2=256, C=24.
// ---------------------------------------------------------------------------

typedef __attribute__((ext_vector_type(16))) __bf16 v16bf;
typedef __attribute__((ext_vector_type(8)))  __bf16 v8bf;
typedef __attribute__((ext_vector_type(8)))  float  v8f;
typedef unsigned short u16;

#define DEV __device__ __forceinline__

DEV u16 f2bf(float f) {
  unsigned u = __builtin_bit_cast(unsigned, f);
  unsigned r = u + 0x7FFFu + ((u >> 16) & 1u);   // round-to-nearest-even
  return (u16)(r >> 16);
}

// A tile: 16x32 bf16, M=lane&15, K-base=(lane>>4)*8; elems [kb..kb+7] then [kb+16..kb+23]
DEV v16bf load_a(const u16* A, int ld, int m0, int k0, int lane) {
  int m  = m0 + (lane & 15);
  int kb = k0 + ((lane >> 4) << 3);
  const u16* p = A + (size_t)m * ld + kb;
  v8bf lo = *(const v8bf*)p;
  v8bf hi = *(const v8bf*)(p + 16);
  return __builtin_shufflevector(lo, hi, 0,1,2,3,4,5,6,7,8,9,10,11,12,13,14,15);
}
// B tile: 32x16 bf16, N=lane&15, K-base=(lane>>4)*8; elems [kb..kb+7] then [kb+16..kb+23]
DEV v16bf load_b(const u16* W, int ld, int n0, int k0, int lane) {
  int n  = n0 + (lane & 15);
  int kb = k0 + ((lane >> 4) << 3);
  const u16* p = W + (size_t)n * ld + kb;
  v8bf lo = *(const v8bf*)p;
  v8bf hi = *(const v8bf*)(p + 16);
  return __builtin_shufflevector(lo, hi, 0,1,2,3,4,5,6,7,8,9,10,11,12,13,14,15);
}

DEV float sigmoidf(float x) { return 1.0f / (1.0f + expf(-x)); }

// ---------------------------------------------------------------------------
// Generic WMMA GEMM: C[128,N] = A[128,K](bf16) @ W[N,K]^T(bf16) + bias
// grid.x = N/16, block = 256 (8 waves = 8 M-tiles of 16)
// ---------------------------------------------------------------------------
__global__ void k_gemm(const u16* __restrict__ A, int lda,
                       const u16* __restrict__ W, int ldw,
                       const float* __restrict__ bias,
                       float* __restrict__ C, int ldc, int K) {
  const int lane  = threadIdx.x & 31;
  const int m0    = (threadIdx.x >> 5) << 4;
  const int n0    = blockIdx.x << 4;
  v8f acc = {};
  for (int k0 = 0; k0 < K; k0 += 32) {
    v16bf a = load_a(A, lda, m0, k0, lane);
    v16bf b = load_b(W, ldw, n0, k0, lane);
    acc = __builtin_amdgcn_wmma_f32_16x16x32_bf16(false, a, false, b, (short)0, acc,
                                                  false, false);
  }
  const int col   = n0 + (lane & 15);
  const int rbase = m0 + ((lane >> 4) << 3);
  const float bv  = bias ? bias[col] : 0.0f;
#pragma unroll
  for (int v = 0; v < 8; ++v)
    C[(size_t)(rbase + v) * ldc + col] = acc[v] + bv;
}

// ---------------------------------------------------------------------------
// Fused GRU-step GEMM: gi = A1@W1^T (+b1)(+Cadd[128,3072]), gh = A2@W2^T + b2
// G[128,4096]: cols 0..3071 = gi+gh ; cols 3072..4095 = gh of n-gate (2048..3071)
// grid = (192, dirs), block = 256
// ---------------------------------------------------------------------------
struct GemmGruArgs {
  const u16* A1; int lda1; const u16* W1; int ldw1; int K1;
  const float* b1; const float* Cadd;
  const u16* A2; int lda2; const u16* W2; const float* b2;
  float* G;
};

__global__ void k_gemm_gru(GemmGruArgs g0, GemmGruArgs g1) {
  const GemmGruArgs g = (blockIdx.y == 0) ? g0 : g1;
  const int lane = threadIdx.x & 31;
  const int m0   = (threadIdx.x >> 5) << 4;
  const int n0   = blockIdx.x << 4;
  v8f acc1 = {}, acc2 = {};
  for (int k0 = 0; k0 < g.K1; k0 += 32) {
    v16bf a = load_a(g.A1, g.lda1, m0, k0, lane);
    v16bf b = load_b(g.W1, g.ldw1, n0, k0, lane);
    acc1 = __builtin_amdgcn_wmma_f32_16x16x32_bf16(false, a, false, b, (short)0, acc1,
                                                   false, false);
  }
  for (int k0 = 0; k0 < 1024; k0 += 32) {
    v16bf a = load_a(g.A2, g.lda2, m0, k0, lane);
    v16bf b = load_b(g.W2, 1024, n0, k0, lane);
    acc2 = __builtin_amdgcn_wmma_f32_16x16x32_bf16(false, a, false, b, (short)0, acc2,
                                                   false, false);
  }
  const int col   = n0 + (lane & 15);
  const int rbase = m0 + ((lane >> 4) << 3);
  const float b1v = g.b1 ? g.b1[col] : 0.0f;
  const float b2v = g.b2[col];
  const bool  isn = (n0 >= 2048);
#pragma unroll
  for (int v = 0; v < 8; ++v) {
    const int row = rbase + v;
    float x1 = acc1[v] + b1v + (g.Cadd ? g.Cadd[(size_t)row * 3072 + col] : 0.0f);
    float x2 = acc2[v] + b2v;
    g.G[(size_t)row * 4096 + col] = x1 + x2;
    if (isn) g.G[(size_t)row * 4096 + col + 1024] = x2;
  }
}

// ---------------------------------------------------------------------------
// GRU gate fusion: r=sig, u=sig, n=tanh(i_n + r*h_n); h' = (1-u)*n + u*h
// grid = (512, dirs), block = 256 (covers 128*1024)
// ---------------------------------------------------------------------------
struct GateArgs {
  const float* G; const float* hprev; float* h; u16* hbf; int hbf_ld;
  float* hcopy; u16* hbfcopy;
};

__global__ void k_gate(GateArgs a0, GateArgs a1) {
  const GateArgs a = (blockIdx.y == 0) ? a0 : a1;
  const int idx = blockIdx.x * blockDim.x + threadIdx.x;
  if (idx >= 128 * 1024) return;
  const int b = idx >> 10, j = idx & 1023;
  const float* Gr = a.G + (size_t)b * 4096;
  float r  = sigmoidf(Gr[j]);
  float u  = sigmoidf(Gr[1024 + j]);
  float s  = Gr[2048 + j];          // i_n + h_n
  float hn = Gr[3072 + j];          // h_n
  float n  = tanhf(s - hn + r * hn);
  float hv = (1.0f - u) * n + u * a.hprev[idx];
  a.h[idx] = hv;
  u16 hb = f2bf(hv);
  a.hbf[(size_t)b * a.hbf_ld + j] = hb;
  if (a.hcopy)   a.hcopy[idx] = hv;
  if (a.hbfcopy) a.hbfcopy[(size_t)b * 1024 + j] = hb;
}

// ---------------------------------------------------------------------------
// Prep / utility kernels
// ---------------------------------------------------------------------------
__global__ void k_zero(float* p, int n) {
  int i = blockIdx.x * blockDim.x + threadIdx.x;
  if (i < n) p[i] = 0.0f;
}

// dst[r][c] (r<drows, c<dld) = (r<srows && c<cols) ? src[r*sld+c] : 0
__global__ void k_convert_pad(u16* dst, int dld, const float* src, int sld,
                              int drows, int srows, int cols) {
  int i = blockIdx.x * blockDim.x + threadIdx.x;
  int total = drows * dld;
  if (i >= total) return;
  int r = i / dld, c = i - r * dld;
  float v = (r < srows && c < cols) ? src[(size_t)r * sld + c] : 0.0f;
  dst[i] = f2bf(v);
}

__global__ void k_pad_bias(float* dst, const float* src, int n, int total) {
  int i = blockIdx.x * blockDim.x + threadIdx.x;
  if (i < total) dst[i] = (i < n) ? src[i] : 0.0f;
}

// Build time-major bf16 x and decoder teacher-forced input, K padded 130->160
__global__ void k_prep_seq(const float* __restrict__ x, u16* __restrict__ xT,
                           u16* __restrict__ decT) {
  int i = blockIdx.x * blockDim.x + threadIdx.x;
  const int total = 256 * 128 * 160;
  if (i >= total) return;
  int t = i / (128 * 160);
  int rem = i - t * (128 * 160);
  int b = rem / 160, c = rem - b * 160;
  float xv = (c < 130) ? x[((size_t)b * 256 + t) * 130 + c] : 0.0f;
  xT[i] = f2bf(xv);
  float dv;
  if (t == 0)       dv = (c == 129) ? 1.0f : 0.0f;       // one-hot start token
  else if (c < 130) dv = x[((size_t)b * 256 + (t - 1)) * 130 + c];
  else              dv = 0.0f;
  decT[i] = f2bf(dv);
}

// mu/std/z outputs + z_cat(bf16, padded 280->288)
__global__ void k_zpack(const float* __restrict__ mu_lin, const float* __restrict__ var_lin,
                        const float* __restrict__ eps, const float* __restrict__ chroma,
                        float* __restrict__ o_mu, float* __restrict__ o_std,
                        float* __restrict__ o_z, u16* __restrict__ zcat) {
  int i = blockIdx.x * blockDim.x + threadIdx.x;
  const int total = 128 * 288;
  if (i >= total) return;
  int b = i / 288, c = i - b * 288;
  float zc = 0.0f;
  if (c < 256) {
    float m = mu_lin[(size_t)b * 256 + c];
    float s = expf(var_lin[(size_t)b * 256 + c]);
    float z = m + s * eps[(size_t)b * 256 + c];
    o_mu[(size_t)b * 256 + c] = m;
    o_std[(size_t)b * 256 + c] = s;
    o_z[(size_t)b * 280 + c] = z;
    zc = z;
  } else if (c < 280) {
    float ch = chroma[(size_t)b * 24 + (c - 256)];
    o_z[(size_t)b * 280 + c] = ch;
    zc = ch;
  }
  zcat[i] = f2bf(zc);
}

// log_softmax over 130 logits per row; one wave per row, 4 rows per block.
__global__ void k_logsoftmax(const float* __restrict__ logits, float* __restrict__ out,
                             int t) {
  const int lane = threadIdx.x & 31;
  const int b = blockIdx.x * 4 + (threadIdx.x >> 5);
  float xs[5];
#pragma unroll
  for (int i = 0; i < 5; ++i) {
    int j = lane + i * 32;
    xs[i] = (j < 130) ? logits[(size_t)b * 144 + j] : -INFINITY;
  }
  float m = xs[0];
#pragma unroll
  for (int i = 1; i < 5; ++i) m = fmaxf(m, xs[i]);
  for (int off = 16; off > 0; off >>= 1) m = fmaxf(m, __shfl_xor(m, off, 32));
  float s = 0.0f;
#pragma unroll
  for (int i = 0; i < 5; ++i) s += expf(xs[i] - m);   // exp(-inf)=0 for padded lanes
  for (int off = 16; off > 0; off >>= 1) s += __shfl_xor(s, off, 32);
  const float ls = m + logf(s);
#pragma unroll
  for (int i = 0; i < 5; ++i) {
    int j = lane + i * 32;
    if (j < 130) out[((size_t)b * 256 + t) * 130 + j] = xs[i] - ls;
  }
}

// ---------------------------------------------------------------------------
// Host orchestration
// ---------------------------------------------------------------------------
extern "C" void kernel_launch(void* const* d_in, const int* in_sizes, int n_in,
                              void* d_out, int out_size, void* d_ws, size_t ws_size,
                              hipStream_t stream) {
  (void)in_sizes; (void)n_in; (void)out_size; (void)ws_size;
  const int T = 256, B = 128, H = 1024;

  const float* x       = (const float*)d_in[0];
  const float* chroma  = (const float*)d_in[1];
  const float* eps     = (const float*)d_in[2];
  const float* w_ih_f  = (const float*)d_in[3];
  const float* w_hh_f  = (const float*)d_in[4];
  const float* b_ih_f  = (const float*)d_in[5];
  const float* b_hh_f  = (const float*)d_in[6];
  const float* w_ih_b  = (const float*)d_in[7];
  const float* w_hh_b  = (const float*)d_in[8];
  const float* b_ih_b  = (const float*)d_in[9];
  const float* b_hh_b  = (const float*)d_in[10];
  const float* w_mu    = (const float*)d_in[11];
  const float* b_mu    = (const float*)d_in[12];
  const float* w_var   = (const float*)d_in[13];
  const float* b_var   = (const float*)d_in[14];
  const float* w_init  = (const float*)d_in[15];
  const float* b_init  = (const float*)d_in[16];
  const float* w_ih_g  = (const float*)d_in[17];
  const float* w_hh_g  = (const float*)d_in[18];
  const float* b_ih_g  = (const float*)d_in[19];
  const float* b_hh_g  = (const float*)d_in[20];
  const float* w_ih_g2 = (const float*)d_in[21];
  const float* w_hh_g2 = (const float*)d_in[22];
  const float* b_ih_g2 = (const float*)d_in[23];
  const float* b_hh_g2 = (const float*)d_in[24];
  const float* w_out   = (const float*)d_in[25];
  const float* b_out   = (const float*)d_in[26];

  float* o_out = (float*)d_out;
  float* o_mu  = o_out + (size_t)B * T * 130;
  float* o_std = o_mu + (size_t)B * 256;
  float* o_z   = o_std + (size_t)B * 256;

  // workspace carve-up (256B aligned)
  char* wp = (char*)d_ws;
  auto alloc = [&](size_t bytes) {
    void* r = (void*)wp;
    wp += (bytes + 255) & ~(size_t)255;
    return r;
  };
  u16*  xT_bf    = (u16*)alloc((size_t)T * B * 160 * 2);
  u16*  decT_bf  = (u16*)alloc((size_t)T * B * 160 * 2);
  u16*  wihf_bf  = (u16*)alloc((size_t)3072 * 160 * 2);
  u16*  wihb_bf  = (u16*)alloc((size_t)3072 * 160 * 2);
  u16*  whhf_bf  = (u16*)alloc((size_t)3072 * 1024 * 2);
  u16*  whhb_bf  = (u16*)alloc((size_t)3072 * 1024 * 2);
  u16*  wmu_bf   = (u16*)alloc((size_t)256 * 2048 * 2);
  u16*  wvar_bf  = (u16*)alloc((size_t)256 * 2048 * 2);
  u16*  winit_bf = (u16*)alloc((size_t)1024 * 288 * 2);
  u16*  wgx_bf   = (u16*)alloc((size_t)3072 * 160 * 2);
  u16*  wgz_bf   = (u16*)alloc((size_t)3072 * 288 * 2);
  u16*  whg_bf   = (u16*)alloc((size_t)3072 * 1024 * 2);
  u16*  wg2i_bf  = (u16*)alloc((size_t)3072 * 1024 * 2);
  u16*  wg2h_bf  = (u16*)alloc((size_t)3072 * 1024 * 2);
  u16*  wout_bf  = (u16*)alloc((size_t)144 * 1024 * 2);
  float* bout_p  = (float*)alloc(144 * 4);
  float* Gf      = (float*)alloc((size_t)B * 4096 * 4);
  float* Gb      = (float*)alloc((size_t)B * 4096 * 4);
  float* G0      = (float*)alloc((size_t)B * 4096 * 4);
  float* G1      = (float*)alloc((size_t)B * 4096 * 4);
  float* hf      = (float*)alloc((size_t)B * H * 4);
  float* hb      = (float*)alloc((size_t)B * H * 4);
  float* h0      = (float*)alloc((size_t)B * H * 4);
  float* h1      = (float*)alloc((size_t)B * H * 4);
  u16*  henc_bf  = (u16*)alloc((size_t)B * 2048 * 2);
  u16*  h0_bf    = (u16*)alloc((size_t)B * H * 2);
  u16*  h1_bf    = (u16*)alloc((size_t)B * H * 2);
  float* mu_lin  = (float*)alloc((size_t)B * 256 * 4);
  float* var_lin = (float*)alloc((size_t)B * 256 * 4);
  float* gz      = (float*)alloc((size_t)B * 3072 * 4);
  u16*  zcat_bf  = (u16*)alloc((size_t)B * 288 * 2);
  float* logits  = (float*)alloc((size_t)B * 144 * 4);

  auto cgrid = [](int n) { return dim3((unsigned)((n + 255) / 256)); };

  // ---- prep: sequence tensors + bf16 weights ----
  k_prep_seq<<<cgrid(T * B * 160), 256, 0, stream>>>(x, xT_bf, decT_bf);
  k_convert_pad<<<cgrid(3072 * 160), 256, 0, stream>>>(wihf_bf, 160, w_ih_f, 130, 3072, 3072, 130);
  k_convert_pad<<<cgrid(3072 * 160), 256, 0, stream>>>(wihb_bf, 160, w_ih_b, 130, 3072, 3072, 130);
  k_convert_pad<<<cgrid(3072 * 1024), 256, 0, stream>>>(whhf_bf, 1024, w_hh_f, 1024, 3072, 3072, 1024);
  k_convert_pad<<<cgrid(3072 * 1024), 256, 0, stream>>>(whhb_bf, 1024, w_hh_b, 1024, 3072, 3072, 1024);
  k_convert_pad<<<cgrid(256 * 2048), 256, 0, stream>>>(wmu_bf, 2048, w_mu, 2048, 256, 256, 2048);
  k_convert_pad<<<cgrid(256 * 2048), 256, 0, stream>>>(wvar_bf, 2048, w_var, 2048, 256, 256, 2048);
  k_convert_pad<<<cgrid(1024 * 288), 256, 0, stream>>>(winit_bf, 288, w_init, 280, 1024, 1024, 280);
  k_convert_pad<<<cgrid(3072 * 160), 256, 0, stream>>>(wgx_bf, 160, w_ih_g, 410, 3072, 3072, 130);
  k_convert_pad<<<cgrid(3072 * 288), 256, 0, stream>>>(wgz_bf, 288, w_ih_g + 130, 410, 3072, 3072, 280);
  k_convert_pad<<<cgrid(3072 * 1024), 256, 0, stream>>>(whg_bf, 1024, w_hh_g, 1024, 3072, 3072, 1024);
  k_convert_pad<<<cgrid(3072 * 1024), 256, 0, stream>>>(wg2i_bf, 1024, w_ih_g2, 1024, 3072, 3072, 1024);
  k_convert_pad<<<cgrid(3072 * 1024), 256, 0, stream>>>(wg2h_bf, 1024, w_hh_g2, 1024, 3072, 3072, 1024);
  k_convert_pad<<<cgrid(144 * 1024), 256, 0, stream>>>(wout_bf, 1024, w_out, 1024, 144, 130, 1024);
  k_pad_bias<<<1, 256, 0, stream>>>(bout_p, b_out, 130, 144);
  k_zero<<<cgrid(B * H), 256, 0, stream>>>(hf, B * H);
  k_zero<<<cgrid(B * H), 256, 0, stream>>>(hb, B * H);
  k_zero<<<cgrid(B * 1024), 256, 0, stream>>>((float*)henc_bf, B * 1024);  // 128*2048 bf16

  // ---- bidirectional GRU encoder: one fused GEMM + gate per step (both dirs) ----
  for (int t = 0; t < T; ++t) {
    GemmGruArgs ef = { xT_bf + (size_t)t * B * 160, 160, wihf_bf, 160, 160,
                       b_ih_f, nullptr, henc_bf, 2048, whhf_bf, b_hh_f, Gf };
    GemmGruArgs eb = { xT_bf + (size_t)(T - 1 - t) * B * 160, 160, wihb_bf, 160, 160,
                       b_ih_b, nullptr, henc_bf + 1024, 2048, whhb_bf, b_hh_b, Gb };
    k_gemm_gru<<<dim3(192, 2), 256, 0, stream>>>(ef, eb);
    GateArgs gf = { Gf, hf, hf, henc_bf, 2048, nullptr, nullptr };
    GateArgs gb = { Gb, hb, hb, henc_bf + 1024, 2048, nullptr, nullptr };
    k_gate<<<dim3(512, 2), 256, 0, stream>>>(gf, gb);
  }

  // ---- heads: mu, std, z, hx0, time-invariant decoder input projection ----
  k_gemm<<<16, 256, 0, stream>>>(henc_bf, 2048, wmu_bf, 2048, b_mu, mu_lin, 256, 2048);
  k_gemm<<<16, 256, 0, stream>>>(henc_bf, 2048, wvar_bf, 2048, b_var, var_lin, 256, 2048);
  k_zpack<<<cgrid(B * 288), 256, 0, stream>>>(mu_lin, var_lin, eps, chroma,
                                              o_mu, o_std, o_z, zcat_bf);
  k_gemm<<<64, 256, 0, stream>>>(zcat_bf, 288, winit_bf, 288, b_init, h0, 1024, 288);
  k_convert_pad<<<cgrid(B * 1024), 256, 0, stream>>>(h0_bf, 1024, h0, 1024, B, B, 1024);
  k_gemm<<<192, 256, 0, stream>>>(zcat_bf, 288, wgz_bf, 288, b_ih_g, gz, 3072, 288);

  // ---- 2-layer autoregressive decoder ----
  for (int t = 0; t < T; ++t) {
    GemmGruArgs d1 = { decT_bf + (size_t)t * B * 160, 160, wgx_bf, 160, 160,
                       nullptr, gz, h0_bf, 1024, whg_bf, b_hh_g, G0 };
    k_gemm_gru<<<dim3(192, 1), 256, 0, stream>>>(d1, d1);
    GateArgs g1a = { G0, h0, h0, h0_bf, 1024,
                     (t == 0) ? h1 : nullptr, (t == 0) ? h1_bf : nullptr };
    k_gate<<<dim3(512, 1), 256, 0, stream>>>(g1a, g1a);

    GemmGruArgs d2 = { h0_bf, 1024, wg2i_bf, 1024, 1024,
                       b_ih_g2, nullptr, h1_bf, 1024, wg2h_bf, b_hh_g2, G1 };
    k_gemm_gru<<<dim3(192, 1), 256, 0, stream>>>(d2, d2);
    GateArgs g2a = { G1, h1, h1, h1_bf, 1024, nullptr, nullptr };
    k_gate<<<dim3(512, 1), 256, 0, stream>>>(g2a, g2a);

    k_gemm<<<9, 256, 0, stream>>>(h1_bf, 1024, wout_bf, 1024, bout_p, logits, 144, 1024);
    k_logsoftmax<<<32, 128, 0, stream>>>(logits, o_out, t);
  }
}
